// SGAT_31250182046267
// MI455X (gfx1250) — matrix operand order repaired and checked by reading, not compiled, not run on hardware
//
#include <hip/hip_runtime.h>
#include <hip/hip_bf16.h>

// ---------------------------------------------------------------------------
// MI455X (gfx1250) fused 3x3 neighborhood attention.
//   Kernel 0: x f32 -> f16 (feeds the TDM, halves repeated x traffic).
//   Kernel 1: qkv = W(384x128) * x(128 x 73728) via v_wmma_f32_16x16x32_f16.
//             X tile staged to LDS by the Tensor Data Mover (tensor_load_to_lds,
//             TENSORcnt), with TDM row padding giving the bank-friendly pitch.
//   Kernel 2: per-head 3x3 window attention with LDS halo tiles.
// ---------------------------------------------------------------------------

typedef _Float16 v16h __attribute__((ext_vector_type(16)));
typedef _Float16 v8h  __attribute__((ext_vector_type(8)));
typedef _Float16 v4h  __attribute__((ext_vector_type(4)));
typedef float    v8f  __attribute__((ext_vector_type(8)));
typedef unsigned int v4u __attribute__((ext_vector_type(4)));
typedef int      v8i  __attribute__((ext_vector_type(8)));
typedef int      v4i  __attribute__((ext_vector_type(4)));

union F16frag { v16h v; v8h h[2]; };

#define BATCH 8
#define CIN   128
#define IMG_H 96
#define IMG_W 96
#define NPIX  (IMG_H * IMG_W)      // 9216
#define OUTC  384                  // 3 * 128 (q,k,v)
#define NHEAD 4
#define HDIM  32

// LDS row pitches (in halfs): 16B-aligned rows, per-lane dword stride has
// gcd(.,64)=4 -> 16 lanes of a fragment load hit 16 distinct bank groups.
#define WPITCH 136   // 128 + 8
#define XPITCH 72    // 64 + 8  (TDM pad: +4 dwords every 32 dwords)

// ---------------------------------------------------------------------------
// Kernel 0: elementwise f32 -> f16 convert of x into workspace.
// ---------------------------------------------------------------------------
__global__ __launch_bounds__(256) void cvt_f32_to_f16(const float* __restrict__ x,
                                                      _Float16* __restrict__ xh) {
    size_t i = ((size_t)blockIdx.x * 256 + threadIdx.x) * 4;
    const float4 v = *(const float4*)(x + i);
    v4h h;
    h.x = (_Float16)v.x; h.y = (_Float16)v.y;
    h.z = (_Float16)v.z; h.w = (_Float16)v.w;
    *(v4h*)(xh + i) = h;
}

// ---------------------------------------------------------------------------
// Kernel 1: one workgroup = 128 output channels (mb) x 64 pixels (nb), K=128.
// 8 waves, each wave owns one 16-row M tile across the four 16-pixel N tiles.
// ---------------------------------------------------------------------------
__global__ __launch_bounds__(256) void qkv_gemm(const _Float16* __restrict__ xh,
                                                const float* __restrict__ Wt,
                                                _Float16* __restrict__ t) {
    __shared__ _Float16 Wl[128 * WPITCH];  // 34816 B
    __shared__ _Float16 Xl[128 * XPITCH];  // 18432 B
    const int tid = threadIdx.x;
    const int nb  = blockIdx.x;            // 0..143 -> 64-pixel strip
    const int mb  = blockIdx.y;            // 0..2   -> 128-channel strip
    const int b   = blockIdx.z;            // 0..7
    const int p0  = nb * 64;
    const int m0g = mb * 128;

    // ---- TDM: async-load the [128 ch][64 px] f16 tile into Xl ([c][n],
    // pitch 72 halfs via TDM LDS padding). Wave 0 issues; TENSORcnt tracks.
    if (tid < 32) {
        const size_t gaddr = (size_t)(xh + (size_t)b * CIN * NPIX + p0);
        const unsigned ga_lo = (unsigned)gaddr;
        const unsigned ga_hi = (unsigned)(gaddr >> 32);
        const unsigned lds_a = (unsigned)(size_t)&Xl[0];  // LDS offset = addr[31:0]

        v4u g0;
        g0.x = 1u;                                  // count=1, user mode
        g0.y = lds_a;                               // lds_addr (bytes)
        g0.z = ga_lo;                               // global_addr[31:0]
        g0.w = (ga_hi & 0x01FFFFFFu) | (2u << 30);  // global_addr[56:32] | type=2

        v8i g1;
        g1[0] = (1 << 16)      // data_size = 1 (2 bytes)
              | (1 << 20)      // pad_enable
              | (4 << 22)      // pad_interval: 1<<4 * 2 = 32 dwords (64 px f16)
              | (3 << 25);     // pad_amount: 4 dwords (8 halfs) -> pitch 72
        g1[1] = (NPIX & 0xFFFF) << 16;              // tensor_dim0[15:0] (=9216)
        g1[2] = (NPIX >> 16) | (CIN << 16);         // tensor_dim0[31:16] | dim1[15:0]
        g1[3] = (64 << 16);                         // tile_dim0 = 64 px
        g1[4] = CIN;                                // tile_dim1 = 128 ch (dim2=0)
        g1[5] = NPIX;                               // tensor_dim0_stride[31:0]
        g1[6] = 0;                                  // stride hi | dim1_stride lo
        g1[7] = 0;

        v4i gz4 = {0, 0, 0, 0};
        v8i gz8 = {0, 0, 0, 0, 0, 0, 0, 0};
        __builtin_amdgcn_tensor_load_to_lds(g0, g1, gz4, gz4, gz8, 0);
        __builtin_amdgcn_s_wait_tensorcnt(0);
    }

    // ---- Stage W chunk [m0g..m0g+128) x [0..128) as f16, [o][c] pitch 136.
#pragma unroll 4
    for (int i = 0; i < 64; ++i) {
        int e = tid + i * 256;
        int c = e & 127, o = e >> 7;
        Wl[o * WPITCH + c] = (_Float16)Wt[(m0g + o) * CIN + c];
    }
    __syncthreads();

    const int wv = tid >> 5, lane = tid & 31;
    const int lm = lane & 15, hi = lane >> 4;  // half-wave select
    const int m0 = wv * 16;                    // wave's M tile inside the 128

    // A fragments (16x32 f16, ISA 7.12.2): lane = M row; halfs 0-7 = K
    // {base..base+7}, halfs 8-15 = K {base+16..}, base = hi*8.
    F16frag a[4];
#pragma unroll
    for (int ks = 0; ks < 4; ++ks) {
        const _Float16* ap = &Wl[(m0 + lm) * WPITCH + ks * 32 + hi * 8];
        a[ks].h[0] = *(const v8h*)ap;          // ds_load_b128
        a[ks].h[1] = *(const v8h*)(ap + 16);   // ds_load_b128
    }

    _Float16* tb = t + (size_t)b * OUTC * NPIX + (size_t)m0g * NPIX + p0;
#pragma unroll
    for (int nt = 0; nt < 4; ++nt) {
        const int n0 = nt * 16;
        // B fragments (32x16 f16): lane = N column; halfs 0-7 = K
        // {base..base+7}, halfs 8-15 = K {base+16..}, base = hi*8.
        F16frag bf[4];
#pragma unroll
        for (int ks = 0; ks < 4; ++ks) {
            const _Float16* bp = &Xl[(ks * 32 + hi * 8) * XPITCH + n0 + lm];
#pragma unroll
            for (int j = 0; j < 8; ++j) {
                bf[ks].v[j] = bp[j * XPITCH];
                bf[ks].v[8 + j] = bp[(16 + j) * XPITCH];
            }
        }
        v8f acc = {0.f, 0.f, 0.f, 0.f, 0.f, 0.f, 0.f, 0.f};
#pragma unroll
        for (int ks = 0; ks < 4; ++ks)
            acc = __builtin_amdgcn_wmma_f32_16x16x32_f16(
                false, a[ks].v, false, bf[ks].v, (short)0, acc, false, false);

        // C layout: VGPR r -> lanes 0-15 (M=r, N=lane), lanes 16-31 (M=8+r).
        _Float16* ob = tb + (size_t)(m0 + hi * 8) * NPIX + n0 + lm;
#pragma unroll
        for (int r = 0; r < 8; ++r)
            ob[(size_t)r * NPIX] = (_Float16)acc[r];
    }
}

// ---------------------------------------------------------------------------
// Kernel 2: 3x3 neighborhood attention. One block = (b, head, 16x16 tile);
// one thread per pixel. k/v 18x18 halo staged in LDS (zero-padded beyond
// image bounds -> logit 0 participates in softmax, matching the reference).
// ---------------------------------------------------------------------------
#define HALO 18
#define HPIX (HALO * HALO)   // 324
#define KPITCH 40            // 32 + 8 halfs -> 80 B rows, 16B aligned

__global__ __launch_bounds__(256) void attn3x3(const _Float16* __restrict__ t,
                                               float* __restrict__ out) {
    __shared__ _Float16 kt[HPIX * KPITCH];   // 25920 B
    __shared__ _Float16 vt[HPIX * KPITCH];   // 25920 B
    const int tid = threadIdx.x;
    const int bx = blockIdx.x, by = blockIdx.y;
    const int g = blockIdx.z & 3, b = blockIdx.z >> 2;

    const _Float16* tb = t + (size_t)b * OUTC * NPIX;
    const _Float16* kb = tb + (size_t)(128 + g * HDIM) * NPIX;
    const _Float16* vb = tb + (size_t)(256 + g * HDIM) * NPIX;
    const int hx0 = bx * 16 - 1, hy0 = by * 16 - 1;

    // Halo fill: 324 px * 32 ch; consecutive threads -> consecutive pixels of
    // one channel (coalesced 2B global reads along rows).
    for (int i = 0; i < 41; ++i) {
        int e = tid + i * 256;
        if (e < HPIX * HDIM) {
            int ch = e / HPIX, pix = e - ch * HPIX;
            int hy = pix / HALO, hx = pix - hy * HALO;
            int gy = hy0 + hy, gx = hx0 + hx;
            bool in = (gy >= 0) & (gy < IMG_H) & (gx >= 0) & (gx < IMG_W);
            size_t gi = (size_t)ch * NPIX + (size_t)gy * IMG_W + gx;
            kt[pix * KPITCH + ch] = in ? kb[gi] : (_Float16)0.f;
            vt[pix * KPITCH + ch] = in ? vb[gi] : (_Float16)0.f;
        }
    }
    __syncthreads();

    const int py = tid >> 4, px = tid & 15;
    const int gy = by * 16 + py, gx = bx * 16 + px;
    const size_t gp = (size_t)gy * IMG_W + gx;

    // q for this pixel (coalesced across the 16-wide row per channel).
    const _Float16* qb = tb + (size_t)(g * HDIM) * NPIX + gp;
    float q[HDIM];
#pragma unroll
    for (int d = 0; d < HDIM; ++d) q[d] = (float)qb[(size_t)d * NPIX];

    // Logits over the 9 neighbors (torch Unfold order: n = dy*3 + dx).
    float att[9];
    float amax = -1e30f;
#pragma unroll
    for (int n = 0; n < 9; ++n) {
        const int hp = (py + n / 3) * HALO + px + (n % 3);
        const _Float16* kr = &kt[hp * KPITCH];
        float s = 0.f;
#pragma unroll
        for (int c8 = 0; c8 < 4; ++c8) {
            v8h kv = *(const v8h*)(kr + c8 * 8);   // ds_load_b128
#pragma unroll
            for (int j = 0; j < 8; ++j) s += q[c8 * 8 + j] * (float)kv[j];
        }
        s *= 0.17677669529663687f;   // 1/sqrt(32)
        att[n] = s;
        amax = fmaxf(amax, s);
    }
    float denom = 0.f;
#pragma unroll
    for (int n = 0; n < 9; ++n) { att[n] = __expf(att[n] - amax); denom += att[n]; }
    const float inv = 1.f / denom;

    float o[HDIM];
#pragma unroll
    for (int d = 0; d < HDIM; ++d) o[d] = 0.f;
#pragma unroll
    for (int n = 0; n < 9; ++n) {
        const int hp = (py + n / 3) * HALO + px + (n % 3);
        const _Float16* vr = &vt[hp * KPITCH];
        const float w = att[n] * inv;
#pragma unroll
        for (int c8 = 0; c8 < 4; ++c8) {
            v8h vv = *(const v8h*)(vr + c8 * 8);   // ds_load_b128
#pragma unroll
            for (int j = 0; j < 8; ++j) o[c8 * 8 + j] += w * (float)vv[j];
        }
    }

    float* ob = out + (size_t)b * CIN * NPIX + (size_t)(g * HDIM) * NPIX + gp;
#pragma unroll
    for (int d = 0; d < HDIM; ++d) ob[(size_t)d * NPIX] = o[d];
}

// ---------------------------------------------------------------------------
extern "C" void kernel_launch(void* const* d_in, const int* in_sizes, int n_in,
                              void* d_out, int out_size, void* d_ws, size_t ws_size,
                              hipStream_t stream) {
    (void)in_sizes; (void)n_in; (void)out_size; (void)ws_size;
    const float* x  = (const float*)d_in[0];   // [8,128,96,96] f32
    const float* Wt = (const float*)d_in[1];   // [384,128] f32
    float* out = (float*)d_out;                // [8,128,96,96] f32

    // Workspace: t (f16, 56,623,104 B) then xh (f16, 18,874,368 B).
    _Float16* t  = (_Float16*)d_ws;
    _Float16* xh = (_Float16*)((char*)d_ws + (size_t)BATCH * OUTC * NPIX * 2);

    cvt_f32_to_f16<<<dim3((BATCH * CIN * NPIX) / 1024), 256, 0, stream>>>(x, xh);
    qkv_gemm<<<dim3(NPIX / 64, 3, BATCH), 256, 0, stream>>>(xh, Wt, t);
    attn3x3<<<dim3(IMG_W / 16, IMG_H / 16, BATCH * NHEAD), 256, 0, stream>>>(t, out);
}
